// TransConvEncoderModule_8375186227605
// MI455X (gfx1250) — matrix operand order repaired and checked by reading, not compiled, not run on hardware
//
#include <hip/hip_runtime.h>

// ---------------------------------------------------------------------------
// Two self-attention conv layers on [8,512,40,100] -> [8,64,40,100].
// All GEMM work on v_wmma_f32_16x16x32_bf16 (wave32). Flash-style attention:
// scores stay in LDS. Fragments fed by 16-byte vector loads; conv staging is
// branchless (mask-AND zero-padding) and double-buffered (1 barrier / K-step).
// ---------------------------------------------------------------------------

typedef __attribute__((ext_vector_type(16))) __bf16 bf16x16;
typedef __attribute__((ext_vector_type(8)))  float  f32x8;

struct FragBits { uint4 lo, hi; };
static_assert(sizeof(FragBits) == 32, "frag bits");

#define NPIX 4000
#define HH 40
#define WW 100
#define BB 8
#define CC 64
#define CHUNK 800
#define NCHUNKS 5

__device__ __forceinline__ float bf2f(__bf16 b) {
  unsigned short h = __builtin_bit_cast(unsigned short, b);
  unsigned u = ((unsigned)h) << 16;
  return __builtin_bit_cast(float, u);
}
__device__ __forceinline__ __bf16 f2bf(float f) {
  unsigned u = __builtin_bit_cast(unsigned, f);
  unsigned r = u + 0x7fffu + ((u >> 16) & 1u);   // round-to-nearest-even
  unsigned short h = (unsigned short)(r >> 16);
  return __builtin_bit_cast(__bf16, h);
}
__device__ __forceinline__ bf16x16 frag_from(uint4 lo, uint4 hi) {
  FragBits t; t.lo = lo; t.hi = hi;
  return __builtin_bit_cast(bf16x16, t);
}
__device__ __forceinline__ bf16x16 load_frag(const __bf16* p) {
  const uint4* q = (const uint4*)p;
  return frag_from(q[0], q[1]);
}
// Unconditional 32B load masked by m (0 or ~0): no exec-mask branches.
__device__ __forceinline__ bf16x16 load_frag_mask(const __bf16* p, unsigned m) {
  const uint4* q = (const uint4*)p;
  uint4 lo = q[0], hi = q[1];
  lo.x &= m; lo.y &= m; lo.z &= m; lo.w &= m;
  hi.x &= m; hi.y &= m; hi.z &= m; hi.w &= m;
  return frag_from(lo, hi);
}
__device__ __forceinline__ f32x8 wmma_bf16(bf16x16 a, bf16x16 b, f32x8 c) {
  return __builtin_amdgcn_wmma_f32_16x16x32_bf16(false, a, false, b,
                                                 (short)0, c, false, false);
}
// A-fragment element index -> K offset within a 32-wide K step.
__device__ __forceinline__ int a_elem_k(int half, int i) {
  return half * 16 + i;
}

// ---------------------------------------------------------------------------
__global__ void k_cvt_bf16(const float* __restrict__ in,
                           __bf16* __restrict__ out, int n) {
  int i = blockIdx.x * blockDim.x + threadIdx.x;
  if (i < n) out[i] = f2bf(in[i]);
}

// Pack OIHW f32 conv weights into per-lane A-fragment order:
// wp[((kstep*4 + cot)*32 + lane)*16 + i],  kstep = pos*(Cin/32)+s
__global__ void k_pack_w(const float* __restrict__ w,   // [64][Cin][9]
                         __bf16* __restrict__ wp, int Cin) {
  int idx = blockIdx.x * blockDim.x + threadIdx.x;
  int nS = Cin >> 5;
  int total = 9 * nS * 4 * 512;
  if (idx >= total) return;
  int i    = idx & 15;
  int lane = (idx >> 4) & 31;
  int t    = (idx >> 9) & 3;
  int rest = idx >> 11;
  int s    = rest % nS;
  int pos  = rest / nS;
  int half = lane >> 4, m = lane & 15;
  int k = a_elem_k(half, i);
  int co = t * 16 + m, ci = s * 32 + k;
  wp[idx] = f2bf(w[((size_t)co * Cin + ci) * 9 + pos]);
}

// Pack q(16)|k(16)|v(64) 1x1 weights ([*,64] f32) into fragment order.
__global__ void k_pack_qkv(const float* __restrict__ qw,
                           const float* __restrict__ kw,
                           const float* __restrict__ vw,
                           __bf16* __restrict__ wp) {
  int idx = blockIdx.x * blockDim.x + threadIdx.x;
  if (idx >= 2 * 6 * 512) return;
  int i    = idx & 15;
  int lane = (idx >> 4) & 31;
  int rest = idx >> 9;
  int t    = rest % 6;
  int st   = rest / 6;
  int half = lane >> 4, m = lane & 15;
  int k = a_elem_k(half, i);
  int row = t * 16 + m, kk = st * 32 + k;
  float v = (row < 16) ? qw[row * 64 + kk]
          : (row < 32) ? kw[(row - 16) * 64 + kk]
                       : vw[(row - 32) * 64 + kk];
  wp[idx] = f2bf(v);
}

// ---------------------------------------------------------------------------
// 3x3 conv (pad 1) as implicit GEMM + BN + ReLU.
// Block = 256 = 8 waves: 2 pixel-tile groups x 4 co-tiles.
// Staging: unconditional clause loads + 64-bit mask AND for halo zeros,
// fragment-order LDS, double-buffered (one barrier per K-step).
__global__ __launch_bounds__(256) void k_conv3x3(
    const __bf16* __restrict__ xin,   // [B][Cin][4000]
    const __bf16* __restrict__ wp,    // packed A fragments (linear in kstep)
    const float* __restrict__ bns, const float* __restrict__ bnb,
    float* __restrict__ outf,         // [B][64][4000] (may be null)
    __bf16* __restrict__ outb,        // [B][64][4000] (may be null)
    int Cin)
{
  __shared__ alignas(16) __bf16 sB[2][2][512];   // [group][buf][frag order]
  const int tid  = threadIdx.x;
  const int lane = tid & 31, half = lane >> 4, n16 = lane & 15;
  const int wv = tid >> 5, g = wv >> 2, cot = wv & 3;
  const int tloc = tid & 127;
  const int pix = tloc & 15, kcq = tloc >> 4;    // staging role: 4 kc each
  const int ptile = blockIdx.x * 2 + g;
  const int b  = ptile / 250;
  const int p0 = (ptile % 250) * 16;
  const int co0 = cot * 16;
  const int nS = Cin >> 5;
  const int lg = __builtin_ctz(nS);
  const int total = 9 * nS;

  const int sp = p0 + pix;
  const int shh = sp / WW, sww = sp % WW;
  // fragment-order LDS slot for this thread's 4 staged elements
  const int dst = (pix + 16 * (kcq >> 2)) * 16 + (kcq & 3) * 4;
  const __bf16* xb = xin + (size_t)b * Cin * NPIX;

  auto stage = [&](int kstep, int buf) {
    int pos = kstep >> lg;
    int s   = kstep & (nS - 1);
    int dh = pos / 3 - 1, dw = pos % 3 - 1;
    int hh = shh + dh, ww2 = sww + dw;
    bool ok = (hh >= 0 && hh < HH && ww2 >= 0 && ww2 < WW);
    int src = ok ? (hh * WW + ww2) : 0;          // clamped, always in-bounds
    unsigned long long m64 = ok ? ~0ull : 0ull;  // value mask (no branches)
    const __bf16* bp = xb + (size_t)((s << 5) + kcq * 4) * NPIX + src;
    union { __bf16 h[4]; unsigned long long u; } tmp;
    #pragma unroll
    for (int j = 0; j < 4; ++j)
      tmp.h[j] = bp[(size_t)j * NPIX];           // unconditional clause loads
    tmp.u &= m64;                                // zero halo via AND
    *(unsigned long long*)&sB[g][buf][dst] = tmp.u;
  };

  stage(0, 0);
  f32x8 acc = {};
  for (int kstep = 0; kstep < total; ++kstep) {
    __syncthreads();
    if (kstep + 1 < total) stage(kstep + 1, (kstep + 1) & 1);
    if (kstep + 2 < total)                       // global_prefetch_b8
      __builtin_prefetch(wp + (((size_t)(kstep + 2) * 4 + cot) * 32 + lane) * 16, 0, 2);
    bf16x16 afrag = load_frag(wp + (((size_t)kstep * 4 + cot) * 32 + lane) * 16);
    bf16x16 bfrag = load_frag(&sB[g][kstep & 1][lane * 16]);
    acc = wmma_bf16(afrag, bfrag, acc);
  }
  #pragma unroll
  for (int r = 0; r < 8; ++r) {
    int c = co0 + r + half * 8;
    int p = p0 + n16;
    float v = acc[r] * bns[c] + bnb[c];
    v = v > 0.f ? v : 0.f;
    size_t idx = ((size_t)(b * CC + c)) * NPIX + p;
    if (outf) outf[idx] = v;
    if (outb) outb[idx] = f2bf(v);
  }
}

// ---------------------------------------------------------------------------
// Fused Q/K/V 1x1 projections: [96,64] x [64,16] GEMM + bias.
// Block = 192 = 6 waves (one per M tile). Activation tile staged with b128
// loads (8 pixels / channel). Q and K written transposed [B][N][16].
__global__ __launch_bounds__(192) void k_qkv(
    const __bf16* __restrict__ act,   // [B][64][4000]
    const __bf16* __restrict__ wp,    // packed fragments (2 steps x 6 tiles)
    const float* __restrict__ qb, const float* __restrict__ kb,
    const float* __restrict__ vb,
    __bf16* __restrict__ qt,          // [B][4000][16]
    __bf16* __restrict__ kt,          // [B][4000][16]
    __bf16* __restrict__ vout)        // [B][64][4000]
{
  __shared__ alignas(16) __bf16 sA[2 * 512];     // fragment order per K step
  const int tid  = threadIdx.x;
  const int lane = tid & 31, half = lane >> 4, n16 = lane & 15;
  const int wv = tid >> 5;                       // m tile 0..5
  const int b  = blockIdx.x / 250;
  const int p0 = (blockIdx.x % 250) * 16;

  if (tid < 128) {                               // 64 ch x 2 eight-pixel runs
    int kk = tid >> 1, ph = tid & 1;
    union { uint4 q; __bf16 h[8]; } tmp;
    tmp.q = *(const uint4*)(act + ((size_t)(b * CC + kk)) * NPIX + p0 + ph * 8);
    int st = kk >> 5, kloc = kk & 31;
    int base = st * 512 + (kloc >> 4) * 256 + (kloc & 15);
    #pragma unroll
    for (int e = 0; e < 8; ++e)
      sA[base + (ph * 8 + e) * 16] = tmp.h[e];
  }
  __syncthreads();

  f32x8 acc = {};
  #pragma unroll
  for (int st = 0; st < 2; ++st) {
    bf16x16 afrag = load_frag(wp + (((size_t)(st * 6 + wv)) * 32 + lane) * 16);
    bf16x16 bfrag = load_frag(&sA[st * 512 + lane * 16]);
    acc = wmma_bf16(afrag, bfrag, acc);
  }
  #pragma unroll
  for (int r = 0; r < 8; ++r) {
    int mrow = wv * 16 + r + half * 8;
    int p = p0 + n16;
    if (mrow < 16) {
      qt[((size_t)(b * NPIX + p)) * 16 + mrow] = f2bf(acc[r] + qb[mrow]);
    } else if (mrow < 32) {
      int d = mrow - 16;
      kt[((size_t)(b * NPIX + p)) * 16 + d] = f2bf(acc[r] + kb[d]);
    } else {
      int c = mrow - 32;
      vout[((size_t)(b * CC + c)) * NPIX + p] = f2bf(acc[r] + vb[c]);
    }
  }
}

// ---------------------------------------------------------------------------
// Flash-style attention: one block per (batch, 16-query block).
// 5 chunks x 800 keys; bf16 scores in LDS; online softmax; A x V^T per chunk.
__global__ __launch_bounds__(128) void k_attn(
    const __bf16* __restrict__ qt,    // [B][4000][16]
    const __bf16* __restrict__ kt,    // [B][4000][16]
    const __bf16* __restrict__ vin,   // [B][64][4000]
    const float* __restrict__ actf,   // [B][64][4000] residual
    const float* __restrict__ gamma,  // [1]
    __bf16* __restrict__ feat)        // [B][64][4000]
{
  __shared__ alignas(16) __bf16 sE[16 * CHUNK];
  __shared__ float sRed[16 * 8];
  __shared__ float sMax[16], sSum[16], sScale[16];
  const int tid  = threadIdx.x;
  const int lane = tid & 31, half = lane >> 4, n16 = lane & 15;
  const int wv = tid >> 5;                       // c tile 0..3
  const int b  = blockIdx.x / 250;
  const int i0 = (blockIdx.x % 250) * 16;
  const int c0 = wv * 16;
  const int row = tid >> 3, sub = tid & 7;
  const uint4 z4 = {0, 0, 0, 0};
  const unsigned kmask = (half == 0) ? ~0u : 0u; // K pad mask (no branches)

  // Q A-fragment: 32B contiguous load (K 0..15), lanes 16..31 (K 16..31) zero-padded
  bf16x16 aq = load_frag_mask(qt + ((size_t)(b * NPIX + i0 + n16)) * 16, kmask);

  if (tid < 16) { sMax[tid] = -3.0e38f; sSum[tid] = 0.f; }
  __syncthreads();

  f32x8 oacc = {};
  for (int ch = 0; ch < NCHUNKS; ++ch) {
    const int chunk0 = ch * CHUNK;
    // 1) energy tiles E[i,j] = q . k (one WMMA / 16 columns)
    for (int lt = wv; lt < CHUNK / 16; lt += 4) {
      int j0 = chunk0 + lt * 16;
      // unconditional load; lanes 16..31 (K pad) masked to zero via v_and
      bf16x16 bk = load_frag_mask(kt + ((size_t)(b * NPIX + j0 + n16)) * 16,
                                  kmask);
      f32x8 z = {};
      f32x8 e = wmma_bf16(aq, bk, z);
      #pragma unroll
      for (int r = 0; r < 8; ++r)
        sE[(r + half * 8) * CHUNK + lt * 16 + n16] = f2bf(e[r]);
    }
    __syncthreads();
    // 2) chunk row max (8 threads / row)
    float lm = -3.0e38f;
    for (int jj = sub; jj < CHUNK; jj += 8)
      lm = fmaxf(lm, bf2f(sE[row * CHUNK + jj]));
    sRed[row * 8 + sub] = lm;
    __syncthreads();
    if (sub == 0) {
      float nm = sMax[row];
      #pragma unroll
      for (int t = 0; t < 8; ++t) nm = fmaxf(nm, sRed[row * 8 + t]);
      sScale[row] = __expf(sMax[row] - nm);
      sMax[row] = nm;
    }
    __syncthreads();
    // 3) rescale running output accumulators
    #pragma unroll
    for (int r = 0; r < 8; ++r) oacc[r] *= sScale[r + half * 8];
    // 4) exp in place + chunk row sums
    float ls = 0.f;
    for (int jj = sub; jj < CHUNK; jj += 8) {
      float p = __expf(bf2f(sE[row * CHUNK + jj]) - sMax[row]);
      sE[row * CHUNK + jj] = f2bf(p);
      ls += p;
    }
    sRed[row * 8 + sub] = ls;
    __syncthreads();
    if (sub == 0) {
      float cs = 0.f;
      #pragma unroll
      for (int t = 0; t < 8; ++t) cs += sRed[row * 8 + t];
      sSum[row] = sSum[row] * sScale[row] + cs;
    }
    // 5) O += A_chunk x V^T  (25 WMMA steps over 800 keys)
    for (int st = 0; st < CHUNK / 32; ++st) {
      int jj0 = st * 32;
      const __bf16* erow = &sE[n16 * CHUNK + jj0 + half * 16];
      bf16x16 ap = frag_from(*(const uint4*)erow, *(const uint4*)(erow + 8));
      bf16x16 bv = load_frag(vin + ((size_t)(b * CC + c0 + n16)) * NPIX
                             + chunk0 + jj0 + half * 16);
      oacc = wmma_bf16(ap, bv, oacc);
    }
    __syncthreads();
  }
  const float g = gamma[0];
  #pragma unroll
  for (int r = 0; r < 8; ++r) {
    int irow = r + half * 8;
    int i = i0 + irow;
    int c = c0 + n16;
    float v = oacc[r] * (1.f / sSum[irow]) * g +
              actf[((size_t)(b * CC + c)) * NPIX + i];
    feat[((size_t)(b * CC + c)) * NPIX + i] = f2bf(v);
  }
}

// ---------------------------------------------------------------------------
extern "C" void kernel_launch(void* const* d_in, const int* in_sizes, int n_in,
                              void* d_out, int out_size, void* d_ws,
                              size_t ws_size, hipStream_t stream) {
  (void)in_sizes; (void)n_in; (void)out_size; (void)ws_size;
  const float* x      = (const float*)d_in[0];
  const float* pre_w0 = (const float*)d_in[1];
  const float* bn1_s0 = (const float*)d_in[2];
  const float* bn1_b0 = (const float*)d_in[3];
  const float* q_w0   = (const float*)d_in[4];
  const float* q_b0   = (const float*)d_in[5];
  const float* k_w0   = (const float*)d_in[6];
  const float* k_b0   = (const float*)d_in[7];
  const float* v_w0   = (const float*)d_in[8];
  const float* v_b0   = (const float*)d_in[9];
  const float* gamma0 = (const float*)d_in[10];
  const float* fin_w0 = (const float*)d_in[11];
  const float* bn2_s0 = (const float*)d_in[12];
  const float* bn2_b0 = (const float*)d_in[13];
  const float* pre_w1 = (const float*)d_in[14];
  const float* bn1_s1 = (const float*)d_in[15];
  const float* bn1_b1 = (const float*)d_in[16];
  const float* q_w1   = (const float*)d_in[17];
  const float* q_b1   = (const float*)d_in[18];
  const float* k_w1   = (const float*)d_in[19];
  const float* k_b1   = (const float*)d_in[20];
  const float* v_w1   = (const float*)d_in[21];
  const float* v_b1   = (const float*)d_in[22];
  const float* gamma1 = (const float*)d_in[23];
  const float* fin_w1 = (const float*)d_in[24];
  const float* bn2_s1 = (const float*)d_in[25];
  const float* bn2_b1 = (const float*)d_in[26];

  char* ws = (char*)d_ws;
  size_t off = 0;
  auto carve = [&](size_t bytes) -> void* {
    off = (off + 255) & ~(size_t)255;
    void* p = ws + off;
    off += bytes;
    return p;
  };
  __bf16* x_bf    = (__bf16*)carve((size_t)BB * 512 * NPIX * 2);
  __bf16* wp_pre0 = (__bf16*)carve((size_t)9 * 16 * 4 * 512 * 2);
  __bf16* wp_fin0 = (__bf16*)carve((size_t)9 * 2 * 4 * 512 * 2);
  __bf16* wp_pre1 = (__bf16*)carve((size_t)9 * 2 * 4 * 512 * 2);
  __bf16* wp_fin1 = (__bf16*)carve((size_t)9 * 2 * 4 * 512 * 2);
  __bf16* wp_qkv0 = (__bf16*)carve((size_t)2 * 6 * 512 * 2);
  __bf16* wp_qkv1 = (__bf16*)carve((size_t)2 * 6 * 512 * 2);
  float*  act_f   = (float*)carve((size_t)BB * CC * NPIX * 4);
  __bf16* act_b   = (__bf16*)carve((size_t)BB * CC * NPIX * 2);
  __bf16* qtb     = (__bf16*)carve((size_t)BB * NPIX * 16 * 2);
  __bf16* ktb     = (__bf16*)carve((size_t)BB * NPIX * 16 * 2);
  __bf16* vbuf    = (__bf16*)carve((size_t)BB * CC * NPIX * 2);
  __bf16* feat_b  = (__bf16*)carve((size_t)BB * CC * NPIX * 2);
  __bf16* y_bf    = (__bf16*)carve((size_t)BB * CC * NPIX * 2);

  k_cvt_bf16<<<(BB * 512 * NPIX + 255) / 256, 256, 0, stream>>>(
      x, x_bf, BB * 512 * NPIX);
  k_pack_w<<<(9 * 16 * 4 * 512 + 255) / 256, 256, 0, stream>>>(pre_w0, wp_pre0, 512);
  k_pack_w<<<(9 * 2 * 4 * 512 + 255) / 256, 256, 0, stream>>>(fin_w0, wp_fin0, 64);
  k_pack_w<<<(9 * 2 * 4 * 512 + 255) / 256, 256, 0, stream>>>(pre_w1, wp_pre1, 64);
  k_pack_w<<<(9 * 2 * 4 * 512 + 255) / 256, 256, 0, stream>>>(fin_w1, wp_fin1, 64);
  k_pack_qkv<<<(2 * 6 * 512 + 255) / 256, 256, 0, stream>>>(q_w0, k_w0, v_w0, wp_qkv0);
  k_pack_qkv<<<(2 * 6 * 512 + 255) / 256, 256, 0, stream>>>(q_w1, k_w1, v_w1, wp_qkv1);

  auto run_layer = [&](const __bf16* xin, int Cin, const __bf16* pw,
                       const float* b1s, const float* b1b,
                       const __bf16* wqkv, const float* qbias,
                       const float* kbias, const float* vbias,
                       const float* gam, const __bf16* fw,
                       const float* b2s, const float* b2b,
                       float* final_f, __bf16* final_b) {
    k_conv3x3<<<1000, 256, 0, stream>>>(xin, pw, b1s, b1b, act_f, act_b, Cin);
    k_qkv<<<2000, 192, 0, stream>>>(act_b, wqkv, qbias, kbias, vbias,
                                    qtb, ktb, vbuf);
    k_attn<<<2000, 128, 0, stream>>>(qtb, ktb, vbuf, act_f, gam, feat_b);
    k_conv3x3<<<1000, 256, 0, stream>>>(feat_b, fw, b2s, b2b,
                                        final_f, final_b, 64);
  };
  run_layer(x_bf, 512, wp_pre0, bn1_s0, bn1_b0, wp_qkv0, q_b0, k_b0, v_b0,
            gamma0, wp_fin0, bn2_s0, bn2_b0, nullptr, y_bf);
  run_layer(y_bf, 64, wp_pre1, bn1_s1, bn1_b1, wp_qkv1, q_b1, k_b1, v_b1,
            gamma1, wp_fin1, bn2_s1, bn2_b1, (float*)d_out, nullptr);
}
